// OffsetAttention_17652315586572
// MI455X (gfx1250) — matrix operand order repaired and checked by reading, not compiled, not run on hardware
//
#include <hip/hip_runtime.h>
#include <math.h>

typedef __attribute__((ext_vector_type(16))) __bf16 v16bf;
typedef __attribute__((ext_vector_type(8)))  __bf16 v8bf;
typedef __attribute__((ext_vector_type(8)))  float  v8f;

#define NB   16
#define SEQ  2048
#define DIM  128
#define HID  32
#define ROWS (NB*SEQ)     // 32768
#define RT   (ROWS/16)    // 2048 row tiles
#define MT   (SEQ/16)     // 128 column tiles per batch

// ---------------------------------------------------------------- WMMA helpers

static __device__ __forceinline__ v8f wmma_bf16(v16bf a, v16bf b, v8f c) {
  // D = A(16x32 bf16) * B(32x16 bf16) + C(16x16 f32)
  return __builtin_amdgcn_wmma_f32_16x16x32_bf16(false, a, false, b, (short)0, c,
                                                 false, false);
}

// A fragment (16x32 bf16, row-major source). Lanes 0-15: K {k0..k0+15}
// Lanes 16-31: K {k0+16..k0+31}. Row = row0 + lane%16.
static __device__ __forceinline__ v16bf load_a_frag(const __bf16* base, int stride,
                                                    int row0, int k0) {
  const int lane = threadIdx.x & 31;
  const int lp = lane & 15, half = lane >> 4;
  const __bf16* p = base + (size_t)(row0 + lp) * stride + k0;
  v8bf lo = *reinterpret_cast<const v8bf*>(p + half * 16);
  v8bf hi = *reinterpret_cast<const v8bf*>(p + half * 16 + 8);
  v16bf a;
#pragma unroll
  for (int i = 0; i < 8; ++i) { a[i] = lo[i]; a[8 + i] = hi[i]; }
  return a;
}

// B fragment (32x16 bf16). Source layout: base[col][k] (K contiguous per column).
// Lane col = col0 + lane%16; elements = K k0+half*16 .. +15 (contiguous 32B).
static __device__ __forceinline__ v16bf load_b_frag(const __bf16* base, int stride,
                                                    int col0, int k0) {
  const int lane = threadIdx.x & 31;
  const int lp = lane & 15, half = lane >> 4;
  const __bf16* p = base + (size_t)(col0 + lp) * stride + k0 + half * 16;
  return *reinterpret_cast<const v16bf*>(p);
}

// async global->LDS copy of 16B (per active lane); tracked by ASYNCcnt
static __device__ __forceinline__ void async_copy16(unsigned lds_off,
                                                    const void* gaddr) {
  asm volatile("global_load_async_to_lds_b128 %0, %1, off"
               :: "v"(lds_off), "v"((unsigned long long)(size_t)gaddr)
               : "memory");
}
static __device__ __forceinline__ void async_wait0() {
  asm volatile("s_wait_asynccnt 0x0" ::: "memory");
}

// ---------------------------------------------------------------- kernels

// f32 -> bf16 copy of f_in
__global__ __launch_bounds__(256) void k_convert(const float* __restrict__ src,
                                                 __bf16* __restrict__ dst, int n) {
  int i = blockIdx.x * 256 + threadIdx.x;
  if (i < n) dst[i] = (__bf16)src[i];
}

// convert the four weight matrices to bf16 ([out][in] layout kept -> B-frag ready)
__global__ __launch_bounds__(256) void k_prep(
    const float* __restrict__ wq, const float* __restrict__ wk,
    const float* __restrict__ wv, const float* __restrict__ wl,
    __bf16* __restrict__ wqb, __bf16* __restrict__ wkb,
    __bf16* __restrict__ wvb, __bf16* __restrict__ wlb) {
  int i = blockIdx.x * 256 + threadIdx.x;
  if (i < 4096)        wqb[i]         = (__bf16)wq[i];
  else if (i < 8192)   wkb[i - 4096]  = (__bf16)wk[i - 4096];
  else if (i < 24576)  wvb[i - 8192]  = (__bf16)wv[i - 8192];
  else if (i < 40960)  wlb[i - 24576] = (__bf16)wl[i - 24576];
}

// Fused Q/K/V projection. One wave = one 16x16 output tile.
// ct 0-1 -> q (H=32), ct 2-3 -> k, ct 4-11 -> v (stored transposed: vT[b][e][m]).
__global__ __launch_bounds__(256) void k_qkv(
    const __bf16* __restrict__ finb,
    const __bf16* __restrict__ wqb, const __bf16* __restrict__ wkb,
    const __bf16* __restrict__ wvb,
    const float* __restrict__ bq, const float* __restrict__ bk,
    const float* __restrict__ bv,
    __bf16* __restrict__ qb, __bf16* __restrict__ kb, __bf16* __restrict__ vT) {
  int wave = blockIdx.x * 8 + (threadIdx.x >> 5);
  int lane = threadIdx.x & 31;
  int rt = wave / 12, ct = wave % 12;
  const __bf16* W; const float* bias; int kind, colLocal;
  if (ct < 2)      { W = wqb; bias = bq; kind = 0; colLocal = ct * 16; }
  else if (ct < 4) { W = wkb; bias = bk; kind = 1; colLocal = (ct - 2) * 16; }
  else             { W = wvb; bias = bv; kind = 2; colLocal = (ct - 4) * 16; }
  int row0 = rt * 16;
  v8f acc = {};
#pragma unroll
  for (int kk = 0; kk < 4; ++kk) {
    v16bf a = load_a_frag(finb, DIM, row0, kk * 32);
    v16bf b = load_b_frag(W,    DIM, colLocal, kk * 32);
    acc = wmma_bf16(a, b, acc);
  }
  int lp = lane & 15, half = lane >> 4;
  int col = colLocal + lp;
  float bs = bias[col];
#pragma unroll
  for (int r = 0; r < 8; ++r) {
    float val = acc[r] + bs;
    int row = row0 + (half ? 8 + r : r);
    if (kind == 0)      qb[(size_t)row * HID + col] = (__bf16)val;
    else if (kind == 1) kb[(size_t)row * HID + col] = (__bf16)val;
    else {
      int bb = row >> 11, m = row & (SEQ - 1);
      vT[((size_t)(bb * DIM + col)) * SEQ + m] = (__bf16)val;
    }
  }
}

// Pass 1: per-column (softmax over query axis n) online max + sum.
// One wave owns a 16-column block m and streams all n tiles. K=H=32 -> 1 WMMA/tile.
__global__ __launch_bounds__(256) void k_colstats(
    const __bf16* __restrict__ qb, const __bf16* __restrict__ kb,
    float* __restrict__ colmax, float* __restrict__ colsum) {
  int wave = blockIdx.x * 8 + (threadIdx.x >> 5);
  int lane = threadIdx.x & 31;
  int lp = lane & 15, half = lane >> 4;
  int b = wave >> 7, mt = wave & (MT - 1);
  const __bf16* qbb = qb + (size_t)b * SEQ * HID;
  const __bf16* kbb = kb + (size_t)b * SEQ * HID;
  // B fragment (k for this column block) fixed for the whole loop
  v16bf bk = load_b_frag(kbb, HID, mt * 16, 0);
  float cm = -INFINITY, cs = 0.f;
  for (int nt = 0; nt < MT; ++nt) {
    v16bf a = load_a_frag(qbb, HID, nt * 16, 0);
    v8f z = {};
    v8f s = wmma_bf16(a, bk, z);
    float tm = s[0];
#pragma unroll
    for (int r = 1; r < 8; ++r) tm = fmaxf(tm, s[r]);
    tm = fmaxf(tm, __shfl_xor(tm, 16));     // combine row-halves (same column)
    float nm = fmaxf(cm, tm);
    float part = 0.f;
#pragma unroll
    for (int r = 0; r < 8; ++r) part += __expf(s[r] - nm);
    part += __shfl_xor(part, 16);
    cs = cs * __expf(cm - nm) + part;
    cm = nm;
  }
  if (half == 0) {
    colmax[b * SEQ + mt * 16 + lp] = cm;
    colsum[b * SEQ + mt * 16 + lp] = cs;
  }
}

// Pass 2: flash-style row pass. A block's 8 waves share batch b and the same
// m-schedule, so the k-tile (32x32) and vT-tile (128x32) are staged ONCE per
// block per iteration with async global->LDS copies (ASYNCcnt), then all waves
// read B-fragments from LDS. p is re-laid-out via per-wave LDS transpose.
__global__ __launch_bounds__(256) void k_rowpass(
    const __bf16* __restrict__ qb, const __bf16* __restrict__ kb,
    const __bf16* __restrict__ vT,
    const float* __restrict__ colmax, const float* __restrict__ colsum,
    const float* __restrict__ fin, float* __restrict__ fsa,
    __bf16* __restrict__ xb) {
  __shared__ __attribute__((aligned(32))) __bf16 kbuf[32 * 32];    // 2 KB
  __shared__ __attribute__((aligned(32))) __bf16 vbuf[128 * 32];   // 8 KB
  __shared__ __attribute__((aligned(32))) __bf16 pbuf[8][16 * 32]; // 8 KB
  int wave = blockIdx.x * 8 + (threadIdx.x >> 5);
  int lane = threadIdx.x & 31;
  int lp = lane & 15, half = lane >> 4;
  int b = wave >> 7, nt = wave & (MT - 1);
  __bf16* pb = pbuf[threadIdx.x >> 5];

  const __bf16* qbb = qb + (size_t)b * SEQ * HID;
  const __bf16* kbb = kb + (size_t)b * SEQ * HID;
  const __bf16* vTb = vT + (size_t)b * DIM * SEQ;
  const float* cmp = colmax + b * SEQ;
  const float* csp = colsum + b * SEQ;

  v16bf qa = load_a_frag(qbb, HID, nt * 16, 0);  // fixed A (q rows)
  v8f zero = {};
  v8f acc[8];
#pragma unroll
  for (int e = 0; e < 8; ++e) acc[e] = zero;
  float rs[8];
#pragma unroll
  for (int r = 0; r < 8; ++r) rs[r] = 0.f;

  const int tid = threadIdx.x;
  for (int m0 = 0; m0 < SEQ; m0 += 32) {
    // ---- cooperative async staging of k and vT tiles into LDS ----
    if (tid < 128) {                       // k tile: 128 x 16B chunks
      int mloc = tid >> 2, part = tid & 3;
      async_copy16((unsigned)(size_t)(kbuf + mloc * 32 + part * 8),
                   kbb + (size_t)(m0 + mloc) * HID + part * 8);
    }
#pragma unroll
    for (int c0 = 0; c0 < 2; ++c0) {       // vT tile: 512 x 16B chunks
      int c = tid + c0 * 256;
      int e = c >> 2, part = c & 3;
      async_copy16((unsigned)(size_t)(vbuf + e * 32 + part * 8),
                   vTb + (size_t)e * SEQ + m0 + part * 8);
    }
    async_wait0();
    __syncthreads();

    // ---- scores for the two 16-column sub-tiles ----
#pragma unroll
    for (int tt = 0; tt < 2; ++tt) {
      int mt0 = m0 + tt * 16;
      v16bf bkf = *reinterpret_cast<const v16bf*>(
          kbuf + (tt * 16 + lp) * 32 + half * 16);
      v8f s = wmma_bf16(qa, bkf, zero);
      float cmv  = cmp[mt0 + lp];
      float rinv = 1.0f / csp[mt0 + lp];
#pragma unroll
      for (int r = 0; r < 8; ++r) {
        float p = __expf(s[r] - cmv) * rinv;   // column-softmax'd prob
        rs[r] += p;                            // partial row sum (this lane's col)
        pb[(half ? 8 + r : r) * 32 + tt * 16 + lp] = (__bf16)p;
      }
    }
    // re-read p tile in A-fragment layout (same-wave LDS, in-order)
    v8bf lo = *reinterpret_cast<const v8bf*>(pb + lp * 32 + half * 16);
    v8bf hi = *reinterpret_cast<const v8bf*>(pb + lp * 32 + half * 16 + 8);
    v16bf pa;
#pragma unroll
    for (int i = 0; i < 8; ++i) { pa[i] = lo[i]; pa[8 + i] = hi[i]; }
#pragma unroll
    for (int et = 0; et < 8; ++et) {
      v16bf bvf = *reinterpret_cast<const v16bf*>(
          vbuf + (et * 16 + lp) * 32 + half * 16);
      acc[et] = wmma_bf16(pa, bvf, acc[et]);
    }
    __syncthreads();   // protect k/v tiles from next iteration's staging
  }
  // finish rowsum: reduce over the 16 lanes of each half (each half = same rows)
#pragma unroll
  for (int r = 0; r < 8; ++r) {
    rs[r] += __shfl_xor(rs[r], 1);
    rs[r] += __shfl_xor(rs[r], 2);
    rs[r] += __shfl_xor(rs[r], 4);
    rs[r] += __shfl_xor(rs[r], 8);
    rs[r] = 1.0f / rs[r];
  }
#pragma unroll
  for (int r = 0; r < 8; ++r) {
    int n = nt * 16 + (half ? 8 + r : r);
    size_t row = (size_t)b * SEQ + n;
#pragma unroll
    for (int et = 0; et < 8; ++et) {
      int e = et * 16 + lp;
      float f = acc[et][r] * rs[r];            // f_sa
      fsa[row * DIM + e] = f;
      xb[row * DIM + e] = (__bf16)(fin[row * DIM + e] - f);  // x = f_in - f_sa
    }
  }
}

// y = x @ w_l^T + b_l ; also emit deterministic per-(row-tile, channel) BN partials
__global__ __launch_bounds__(256) void k_linear(
    const __bf16* __restrict__ xb, const __bf16* __restrict__ wlb,
    const float* __restrict__ bl, float* __restrict__ y,
    float* __restrict__ psum, float* __restrict__ psq) {
  int wave = blockIdx.x * 8 + (threadIdx.x >> 5);
  int lane = threadIdx.x & 31;
  int lp = lane & 15, half = lane >> 4;
  int rt = wave >> 3, et = wave & 7;
  int row0 = rt * 16;
  v8f acc = {};
#pragma unroll
  for (int kk = 0; kk < 4; ++kk) {
    v16bf a = load_a_frag(xb,  DIM, row0, kk * 32);
    v16bf b = load_b_frag(wlb, DIM, et * 16, kk * 32);
    acc = wmma_bf16(a, b, acc);
  }
  int e = et * 16 + lp;
  float bias = bl[e];
  float sy = 0.f, sq = 0.f;
#pragma unroll
  for (int r = 0; r < 8; ++r) {
    float v = acc[r] + bias;
    sy += v; sq += v * v;
    y[(size_t)(row0 + (half ? 8 + r : r)) * DIM + e] = v;
  }
  sy += __shfl_xor(sy, 16);
  sq += __shfl_xor(sq, 16);
  if (half == 0) {                 // partials, no float atomics -> deterministic
    psum[(size_t)e * RT + rt] = sy;
    psq [(size_t)e * RT + rt] = sq;
  }
}

// reduce BN partials per channel; produce fused scale/shift
__global__ __launch_bounds__(256) void k_bnstats(
    const float* __restrict__ psum, const float* __restrict__ psq,
    const float* __restrict__ gamma, const float* __restrict__ beta,
    float* __restrict__ bnsc, float* __restrict__ bnsh) {
  __shared__ float ssum[256], ssq[256];
  int e = blockIdx.x;
  const float* p0 = psum + (size_t)e * RT;
  const float* p1 = psq  + (size_t)e * RT;
  float s = 0.f, q = 0.f;
  for (int i = threadIdx.x; i < RT; i += 256) { s += p0[i]; q += p1[i]; }
  ssum[threadIdx.x] = s; ssq[threadIdx.x] = q;
  __syncthreads();
  for (int off = 128; off > 0; off >>= 1) {
    if (threadIdx.x < off) {
      ssum[threadIdx.x] += ssum[threadIdx.x + off];
      ssq [threadIdx.x] += ssq [threadIdx.x + off];
    }
    __syncthreads();
  }
  if (threadIdx.x == 0) {
    float cnt = (float)ROWS;
    float mean = ssum[0] / cnt;
    float var  = ssq[0] / cnt - mean * mean;   // biased variance
    float inv  = rsqrtf(var + 1e-5f);
    float sc   = gamma[e] * inv;
    bnsc[e] = sc;
    bnsh[e] = beta[e] - mean * sc;
  }
}

// out = relu(y*scale + shift) + f_sa
__global__ __launch_bounds__(256) void k_out(
    const float* __restrict__ y, const float* __restrict__ fsa,
    const float* __restrict__ bnsc, const float* __restrict__ bnsh,
    float* __restrict__ out) {
  int i = blockIdx.x * 256 + threadIdx.x;
  int e = i & (DIM - 1);
  float v = y[i] * bnsc[e] + bnsh[e];
  out[i] = fmaxf(v, 0.f) + fsa[i];
}

// ---------------------------------------------------------------- launch

extern "C" void kernel_launch(void* const* d_in, const int* in_sizes, int n_in,
                              void* d_out, int out_size, void* d_ws, size_t ws_size,
                              hipStream_t stream) {
  const float* f_in = (const float*)d_in[0];
  const float* w_q  = (const float*)d_in[1];
  const float* b_q  = (const float*)d_in[2];
  const float* w_k  = (const float*)d_in[3];
  const float* b_k  = (const float*)d_in[4];
  const float* w_v  = (const float*)d_in[5];
  const float* b_v  = (const float*)d_in[6];
  const float* w_l  = (const float*)d_in[7];
  const float* b_l  = (const float*)d_in[8];
  const float* gmm  = (const float*)d_in[9];
  const float* bta  = (const float*)d_in[10];
  float* out = (float*)d_out;

  char* ws = (char*)d_ws;
  size_t off = 0;
  auto take = [&](size_t bytes) -> char* {
    char* p = ws + off;
    off += (bytes + 255) & ~(size_t)255;
    return p;
  };
  __bf16* fin_bf = (__bf16*)take((size_t)ROWS * DIM * 2);
  __bf16* q_bf   = (__bf16*)take((size_t)ROWS * HID * 2);
  __bf16* k_bf   = (__bf16*)take((size_t)ROWS * HID * 2);
  __bf16* vT     = (__bf16*)take((size_t)ROWS * DIM * 2);
  __bf16* x_bf   = (__bf16*)take((size_t)ROWS * DIM * 2);
  __bf16* wq_bf  = (__bf16*)take((size_t)HID * DIM * 2);
  __bf16* wk_bf  = (__bf16*)take((size_t)HID * DIM * 2);
  __bf16* wv_bf  = (__bf16*)take((size_t)DIM * DIM * 2);
  __bf16* wl_bf  = (__bf16*)take((size_t)DIM * DIM * 2);
  float* colmax  = (float*)take((size_t)ROWS * 4);
  float* colsum  = (float*)take((size_t)ROWS * 4);
  float* fsa     = (float*)take((size_t)ROWS * DIM * 4);
  float* ybuf    = (float*)take((size_t)ROWS * DIM * 4);
  float* psum    = (float*)take((size_t)DIM * RT * 4);
  float* psq     = (float*)take((size_t)DIM * RT * 4);
  float* bnsc    = (float*)take(DIM * 4);
  float* bnsh    = (float*)take(DIM * 4);
  (void)ws_size; (void)in_sizes; (void)n_in; (void)out_size;

  const int nElem = ROWS * DIM;                      // 4,194,304
  k_convert<<<nElem / 256, 256, 0, stream>>>(f_in, fin_bf, nElem);
  k_prep<<<160, 256, 0, stream>>>(w_q, w_k, w_v, w_l, wq_bf, wk_bf, wv_bf, wl_bf);
  k_qkv<<<(RT * 12) / 8, 256, 0, stream>>>(fin_bf, wq_bf, wk_bf, wv_bf,
                                           b_q, b_k, b_v, q_bf, k_bf, vT);
  k_colstats<<<(NB * MT) / 8, 256, 0, stream>>>(q_bf, k_bf, colmax, colsum);
  k_rowpass<<<(NB * MT) / 8, 256, 0, stream>>>(q_bf, k_bf, vT, colmax, colsum,
                                               f_in, fsa, x_bf);
  k_linear<<<(RT * 8) / 8, 256, 0, stream>>>(x_bf, wl_bf, b_l, ybuf, psum, psq);
  k_bnstats<<<DIM, 256, 0, stream>>>(psum, psq, gmm, bta, bnsc, bnsh);
  k_out<<<nElem / 256, 256, 0, stream>>>(ybuf, fsa, bnsc, bnsh, out);
}
